// SegMerging_80401787781843
// MI455X (gfx1250) — matrix operand order repaired and checked by reading, not compiled, not run on hardware
//
#include <hip/hip_runtime.h>
#include <hip/hip_bf16.h>

typedef __attribute__((ext_vector_type(16))) _Float16 v16h;
typedef __attribute__((ext_vector_type(8)))  float    v8f;

#define LN_EPS 1e-5f

// Problem constants from setup_inputs()
constexpr int B_      = 16;
constexpr int TSD     = 62;
constexpr int SEG     = 513;
constexpr int D       = 256;              // d_model
constexpr int WIN     = 2;                // win_size
constexpr int WD      = WIN * D;          // 512 (LayerNorm / GEMM K dim)
constexpr int BT      = B_ * TSD;         // 992 independent slices
constexpr int MROWS   = (SEG - 1) / WIN + 1; // 257 merged rows per slice
constexpr int MTILE   = 32;               // rows per block (2 WMMA m-subtiles)
constexpr int MTILES  = (MROWS + MTILE - 1) / MTILE; // 9
constexpr int KTILES  = WD / 32;          // 16 k-steps of 32

// ---------------------------------------------------------------------------
// Prep: W [512][256] f32 (row-major K x N)  ->  Wt [256][512] f16 (N-major)
// Makes each lane's B-fragment (16 consecutive K) a contiguous 32B load.
// ---------------------------------------------------------------------------
__global__ void wt_convert_kernel(const float* __restrict__ W,
                                  _Float16* __restrict__ Wt) {
    int idx = blockIdx.x * blockDim.x + threadIdx.x;
    if (idx >= WD * D) return;
    int k = idx / D;
    int n = idx % D;
    Wt[n * WD + k] = (_Float16)W[idx];
}

// ---------------------------------------------------------------------------
// Fused seg-merge gather + LayerNorm + (512 -> 256) GEMM via f16 WMMA.
// Block: 256 threads (8 wave32). Grid: (9 row-tiles, 992 bt slices).
// Each wave computes a 2x2 grid of 16x16 output tiles (64 WMMAs).
// ---------------------------------------------------------------------------
__global__ __launch_bounds__(256) void segmerge_ln_gemm_kernel(
    const float* __restrict__ x,      // [BT][SEG][D]
    const float* __restrict__ gamma,  // [WD]
    const float* __restrict__ beta,   // [WD]
    const _Float16* __restrict__ Wt,  // [D][WD] f16 (transposed)
    const float* __restrict__ bias,   // [D]
    float* __restrict__ out)          // [BT][MROWS][D]
{
    // A staged in WMMA 16-bit A-fragment order:
    // [m-subtile][ktile][lane][16 halves] = 2 * 16KB = 32KB
    __shared__ _Float16 aFrag[2 * KTILES * 32 * 16];

    const int tile = blockIdx.x;      // row-tile: 0..8
    const int bt   = blockIdx.y;      // 0..991
    const int tid  = threadIdx.x;
    const int wave = tid >> 5;
    const int lane = tid & 31;

    // ---------------- Phase 1: gather + LayerNorm + scatter to aFrag -------
    // Each wave handles 4 rows; lane owns K = lane*16 .. lane*16+15.
    const float* xbt   = x + (size_t)bt * SEG * D;
    const int    half  = lane >> 4;         // which merged slot (0/1)
    const int    Kbase = lane * 16;

    for (int rr = 0; rr < 4; ++rr) {
        const int r = wave * 4 + rr;        // row within tile (0..31)
        const int m = tile * MTILE + r;     // merged-row index
        const bool valid = (m < MROWS);

        int seg = (m == 0) ? 0 : (1 + (m - 1) * WIN + half);
        if (!valid) seg = 0;                // safe address, values zeroed below
        const float* src = xbt + (size_t)seg * D + (lane & 15) * 16;

        float vals[16];
        #pragma unroll
        for (int i = 0; i < 4; ++i) {
            float4 v = *(const float4*)(src + i * 4);
            vals[i * 4 + 0] = v.x; vals[i * 4 + 1] = v.y;
            vals[i * 4 + 2] = v.z; vals[i * 4 + 3] = v.w;
        }
        float s = 0.f, s2 = 0.f;
        #pragma unroll
        for (int i = 0; i < 16; ++i) { s += vals[i]; s2 += vals[i] * vals[i]; }
        // wave32 reduction
        #pragma unroll
        for (int off = 16; off >= 1; off >>= 1) {
            s  += __shfl_xor(s,  off);
            s2 += __shfl_xor(s2, off);
        }
        const float mu  = s * (1.0f / WD);
        const float var = s2 * (1.0f / WD) - mu * mu;
        const float rs  = rsqrtf(var + LN_EPS);

        const int sub  = r >> 4;            // which 16-row m-subtile
        const int rloc = r & 15;
        #pragma unroll
        for (int i = 0; i < 16; ++i) {
            const int K = Kbase + i;
            float yv = (vals[i] - mu) * rs * gamma[K] + beta[K];
            if (!valid) yv = 0.f;
            // A fragment in this kernel's contiguous k order: lane = M + 16*hi,
            // element j holds K%32 = hi*16 + j (matches B fragment)
            const int kt = K >> 5, kk = K & 31;
            const int hi = kk >> 4;
            const int j  = kk & 15;
            aFrag[((sub * KTILES + kt) * 32 + (rloc + 16 * hi)) * 16 + j] =
                (_Float16)yv;
        }
    }
    __syncthreads();

    // ---------------- Phase 2: GEMM via v_wmma_f32_16x16x32_f16 ------------
    // Wave w computes N-tiles w and w+8 (16 cols each) for both m-subtiles.
    const int ncol = lane & 15;
    const int hi   = lane >> 4;
    const int n0   = wave * 16;
    const int n1   = (wave + 8) * 16;
    // B-fragment: lane encodes N; element j holds K = k0 + hi*16 + j
    const _Float16* wp0 = Wt + (size_t)(n0 + ncol) * WD + hi * 16;
    const _Float16* wp1 = Wt + (size_t)(n1 + ncol) * WD + hi * 16;

    v8f acc00 = {}, acc01 = {}, acc10 = {}, acc11 = {};
    #pragma unroll
    for (int kt = 0; kt < KTILES; ++kt) {
        v16h a0 = *(const v16h*)(aFrag + ((0 * KTILES + kt) * 32 + lane) * 16);
        v16h a1 = *(const v16h*)(aFrag + ((1 * KTILES + kt) * 32 + lane) * 16);
        v16h b0 = *(const v16h*)(wp0 + kt * 32);
        v16h b1 = *(const v16h*)(wp1 + kt * 32);
        acc00 = __builtin_amdgcn_wmma_f32_16x16x32_f16(
            false, a0, false, b0, (short)0, acc00, false, false);
        acc01 = __builtin_amdgcn_wmma_f32_16x16x32_f16(
            false, a0, false, b1, (short)0, acc01, false, false);
        acc10 = __builtin_amdgcn_wmma_f32_16x16x32_f16(
            false, a1, false, b0, (short)0, acc10, false, false);
        acc11 = __builtin_amdgcn_wmma_f32_16x16x32_f16(
            false, a1, false, b1, (short)0, acc11, false, false);
    }

    // ---------------- Store: C/D layout M = v + 8*hi, N = lane&15 ----------
    const float bias0 = bias[n0 + ncol];
    const float bias1 = bias[n1 + ncol];
    float* obase = out + (size_t)bt * MROWS * D;
    #pragma unroll
    for (int sub = 0; sub < 2; ++sub) {
        const v8f& a0 = sub ? acc10 : acc00;
        const v8f& a1 = sub ? acc11 : acc01;
        #pragma unroll
        for (int v = 0; v < 8; ++v) {
            const int mrow = tile * MTILE + sub * 16 + v + 8 * hi;
            if (mrow < MROWS) {
                obase[(size_t)mrow * D + n0 + ncol] = a0[v] + bias0;
                obase[(size_t)mrow * D + n1 + ncol] = a1[v] + bias1;
            }
        }
    }
}

extern "C" void kernel_launch(void* const* d_in, const int* in_sizes, int n_in,
                              void* d_out, int out_size, void* d_ws, size_t ws_size,
                              hipStream_t stream) {
    const float* x     = (const float*)d_in[0];
    const float* gamma = (const float*)d_in[1];
    const float* beta  = (const float*)d_in[2];
    const float* W     = (const float*)d_in[3];
    const float* bias  = (const float*)d_in[4];
    float* out         = (float*)d_out;
    _Float16* Wt       = (_Float16*)d_ws;   // 512*256*2 = 256 KB scratch

    // 1) convert + transpose W to f16 (deterministic, rewritten every call)
    wt_convert_kernel<<<(WD * D + 255) / 256, 256, 0, stream>>>(W, Wt);

    // 2) fused gather + LayerNorm + WMMA GEMM
    dim3 grid(MTILES, BT);
    segmerge_ln_gemm_kernel<<<grid, 256, 0, stream>>>(x, gamma, beta, Wt, bias, out);
}
